// BoundedNeuralMemory_91070486545123
// MI455X (gfx1250) — matrix operand order repaired and checked by reading, not compiled, not run on hardware
//
#include <hip/hip_runtime.h>
#include <hip/hip_bf16.h>

#define B_Q   2048
#define N_K   100000
#define DIM   512
#define V_DIM 512
#define TOPK  32
#define TQ    16      // queries per block (one WMMA M tile)
#define CH    128     // keys per chunk (8 waves x 16-key N tiles)
#define KB    128     // K depth per TDM tile
#define A_LD  (DIM + 16)   // 528 bf16: 32B-aligned rows, staggered banks
#define B_LD  (KB  + 16)   // 144 bf16 (= TDM pad: 8 DWORDs per 64 DWORDs)
#define S_LD  132          // padded f32 sims row
#define NSTG  ((N_K + CH - 1) / CH * (DIM / KB))   // 782 chunks * 4 = 3128 stages

typedef __attribute__((ext_vector_type(16))) __bf16 v16bf;
typedef __attribute__((ext_vector_type(8)))  __bf16 v8bf;
typedef __attribute__((ext_vector_type(8)))  float  v8f;
typedef __attribute__((ext_vector_type(4)))  unsigned int u32x4;
typedef __attribute__((ext_vector_type(8)))  int i32x8;
typedef __attribute__((ext_vector_type(4)))  int i32x4;

union AFrag { v16bf v; v8bf h[2]; };

// ---------------- row L2-normalize + bf16 convert ----------------
__global__ __launch_bounds__(256) void bnm_normalize_bf16(
    const float* __restrict__ src, __hip_bfloat16* __restrict__ dst) {
  const int row = blockIdx.x;
  const int t = threadIdx.x;
  const float* s = src + (size_t)row * DIM;
  float v0 = s[t], v1 = s[t + 256];
  float ss = v0 * v0 + v1 * v1;
#pragma unroll
  for (int o = 16; o > 0; o >>= 1) ss += __shfl_xor(ss, o, 32);
  __shared__ float red[8];
  if ((t & 31) == 0) red[t >> 5] = ss;
  __syncthreads();
  float tot = 0.f;
#pragma unroll
  for (int i = 0; i < 8; ++i) tot += red[i];
  const float scale = 1.0f / fmaxf(sqrtf(tot), 1e-8f);
  dst[(size_t)row * DIM + t]       = __float2bfloat16(v0 * scale);
  dst[(size_t)row * DIM + t + 256] = __float2bfloat16(v1 * scale);
}

// wave-cooperative sorted top-k: one list slot per lane, descending order.
__device__ __forceinline__ void topk_insert(unsigned cand, float v, int key,
                                            float& tval, int& tidx,
                                            float& thr, int lane) {
  unsigned mask = cand;
  while (mask) {
    int src = __builtin_ctz(mask);
    mask &= mask - 1;
    float cv = __shfl(v, src, 32);
    int   ck = __shfl(key, src, 32);
    if (cv > thr) {
      unsigned ge = __builtin_amdgcn_ballot_w32(tval >= cv);
      int pos = __builtin_popcount(ge);
      float sv = __shfl_up(tval, 1, 32);
      int   si = __shfl_up(tidx, 1, 32);
      if (lane > pos)       { tval = sv; tidx = si; }
      else if (lane == pos) { tval = cv; tidx = ck; }
      thr = __shfl(tval, 31, 32);
    }
  }
}

// Issue a TDM load of a CH x KB bf16 tile of kn into LDS.
// D# per ISA ch.8: group0 = count/lds_addr/global_addr/type=2
// group1 = data_size=2B, pad 8 DWORDs per 64 DWORDs (-> 288B LDS row stride),
//          tensor dims (dim1 = rows remaining -> hardware zero-fill of OOB rows),
//          tile 128x128, strides = DIM.
__device__ __forceinline__ void tdm_load_tile(const __hip_bfloat16* gsrc,
                                              unsigned ldsAddr, int rowsAvail) {
  unsigned long long ga = (unsigned long long)(uintptr_t)gsrc;
  u32x4 g0 = { 1u,                                     // count=1
               ldsAddr,                                // lds_addr
               (unsigned)(ga & 0xFFFFFFFFull),         // global_addr[31:0]
               ((unsigned)((ga >> 32) & 0x01FFFFFFull)) | 0x80000000u }; // type=2
  i32x8 g1;
  g1[0] = (1 << 16) | (1 << 20) | (5 << 22) | (7 << 25); // data_size=2B,pad_en,int=64dw,amt=8dw
  g1[1] = (DIM & 0xFFFF) << 16;                          // tensor_dim0[15:0]
  g1[2] = ((unsigned)rowsAvail & 0xFFFF) << 16;          // dim0 hi=0 | tensor_dim1[15:0]
  g1[3] = (((unsigned)rowsAvail >> 16) & 0xFFFF) | (KB << 16); // dim1 hi | tile_dim0=128
  g1[4] = CH;                                            // tile_dim1=128, tile_dim2=0
  g1[5] = DIM;                                           // tensor_dim0_stride[31:0]
  g1[6] = (DIM & 0xFFFF) << 16;                          // d0str hi=0 | tensor_dim1_stride[15:0]
  g1[7] = 0;                                             // tensor_dim1_stride hi
  i32x4 z4 = { 0, 0, 0, 0 };
  i32x8 z8 = { 0, 0, 0, 0, 0, 0, 0, 0 };
  __builtin_amdgcn_tensor_load_to_lds(g0, g1, z4, z4, z8, 0);
}

// ------------- fused bf16-WMMA GEMM + TDM double-buffer + top-k + gather -------------
__global__ __launch_bounds__(256) void bnm_retrieve(
    const __hip_bfloat16* __restrict__ qn,
    const __hip_bfloat16* __restrict__ kn,
    const float* __restrict__ values,
    float* __restrict__ out_combined,
    float* __restrict__ out_topk) {

  __shared__ unsigned short sA[TQ * A_LD];      // 16896 B : query tile
  __shared__ unsigned short sB[2][CH * B_LD];   // 2x36864 B : double-buffered key tiles
  __shared__ float          sS[TQ * S_LD];      //  8448 B : sims tile / reused idx+wt

  const int tid  = threadIdx.x;
  const int lane = tid & 31;
  const int w    = tid >> 5;
  const int qBase = blockIdx.x * TQ;

  const unsigned sBaddr[2] = { (unsigned)(uintptr_t)&sB[0][0],
                               (unsigned)(uintptr_t)&sB[1][0] };

  // stage normalized query tile (coalesced uint4)
  {
    const uint4* g = reinterpret_cast<const uint4*>(qn + (size_t)qBase * DIM);
    for (int i = tid; i < TQ * (DIM / 8); i += 256) {
      int r = i >> 6, c = i & 63;
      *reinterpret_cast<uint4*>(&sA[r * A_LD + c * 8]) = g[r * (DIM / 8) + c];
    }
  }

  // per-wave top-k register lists for local queries 2w, 2w+1
  float tv0 = -__builtin_inff(), tv1 = -__builtin_inff();
  int   ti0 = 0, ti1 = 0;
  float th0 = -__builtin_inff(), th1 = -__builtin_inff();

  const int hi   = lane >> 4;
  const int lo16 = lane & 15;
  // A 16-bit 16x32 layout: per lane K = {hi*16..+15} (contiguous, matches B)
  const unsigned short* aLo = &sA[lo16 * A_LD + hi * 16];
  const unsigned short* aHi = aLo + 8;
  const int bOff = (w * 16 + lo16) * B_LD + hi * 16;   // B: 16 contiguous K per lane

  // prologue: kick off stage 0 DMA
  if (w == 0) {
    tdm_load_tile(kn, sBaddr[0], N_K);
  }

  v8f acc = {};
  for (int s = 0; s < NSTG; ++s) {
    if (w == 0) {
      if (s + 1 < NSTG) {
        const int nc = ((s + 1) >> 2) * CH;
        const int nk = ((s + 1) & 3) * KB;
        tdm_load_tile(kn + (size_t)nc * DIM + nk, sBaddr[(s + 1) & 1], N_K - nc);
        __builtin_amdgcn_s_wait_tensorcnt(1);   // stage s tile resident
      } else {
        __builtin_amdgcn_s_wait_tensorcnt(0);
      }
    }
    __syncthreads();                            // all waves: buffer s&1 ready

    const unsigned short* bRow = &sB[s & 1][bOff];
    const int kb = (s & 3) * KB;
#pragma unroll
    for (int kk = 0; kk < KB; kk += 32) {
      AFrag a;
      a.h[0] = *reinterpret_cast<const v8bf*>(aLo + kb + kk);
      a.h[1] = *reinterpret_cast<const v8bf*>(aHi + kb + kk);
      v16bf b = *reinterpret_cast<const v16bf*>(bRow + kk);
      acc = __builtin_amdgcn_wmma_f32_16x16x32_bf16(
          false, a.v, false, b, (short)0, acc, false, false);
    }

    if ((s & 3) == 3) {                         // full K accumulated for this chunk
      const int chunk = (s >> 2) * CH;
      // scatter 16x16 f32 tile: VGPR j -> M=j+hi*8, N=lane%16
#pragma unroll
      for (int j = 0; j < 8; ++j)
        sS[(j + hi * 8) * S_LD + (w * 16 + lo16)] = acc[j];
      __syncthreads();
      // streaming top-k: wave w scans local queries 2w, 2w+1
#pragma unroll
      for (int which = 0; which < 2; ++which) {
        int q = (w << 1) | which;
        float& tv = which ? tv1 : tv0;
        int&   ti = which ? ti1 : ti0;
        float& th = which ? th1 : th0;
#pragma unroll
        for (int base = 0; base < CH; base += 32) {
          int key = chunk + base + lane;
          float v = (key < N_K) ? sS[q * S_LD + base + lane] : -__builtin_inff();
          unsigned cand = __builtin_amdgcn_ballot_w32(v > th);
          if (cand) topk_insert(cand, v, key, tv, ti, th, lane);
        }
      }
      acc = (v8f){};                            // reset for next chunk
    }
    __syncthreads();                            // protects sB reuse (TDM s+2) and sS
  }

  // finalize: top-k values out, softmax weights + indices into reused LDS
  int*   sIdx = reinterpret_cast<int*>(sS);
  float* sWt  = sS + TQ * TOPK;
#pragma unroll
  for (int which = 0; which < 2; ++which) {
    int q = (w << 1) | which;
    float val = which ? tv1 : tv0;
    int   idx = which ? ti1 : ti0;
    out_topk[(size_t)(qBase + q) * TOPK + lane] = val;   // sorted descending
    float m = __shfl(val, 0, 32);
    float e = __expf(val - m);
    float sum = e;
#pragma unroll
    for (int o = 16; o > 0; o >>= 1) sum += __shfl_xor(sum, o, 32);
    sIdx[q * TOPK + lane] = idx;
    sWt[q * TOPK + lane]  = e / sum;
  }
  __syncthreads();

  // weighted gather of value rows: thread owns 2 of 512 output columns
  for (int q = 0; q < TQ; ++q) {
    float a0 = 0.f, a1 = 0.f;
#pragma unroll 4
    for (int j = 0; j < TOPK; ++j) {
      int key  = sIdx[q * TOPK + j];
      float wt = sWt[q * TOPK + j];
      const float* vr = values + (size_t)key * V_DIM;
      a0 += wt * vr[tid];
      a1 += wt * vr[tid + 256];
    }
    float* o = out_combined + (size_t)(qBase + q) * V_DIM;
    o[tid]       = a0;
    o[tid + 256] = a1;
  }
}

extern "C" void kernel_launch(void* const* d_in, const int* in_sizes, int n_in,
                              void* d_out, int out_size, void* d_ws, size_t ws_size,
                              hipStream_t stream) {
  (void)in_sizes; (void)n_in; (void)out_size; (void)ws_size;
  const float* queries = (const float*)d_in[0];
  const float* keys    = (const float*)d_in[1];
  const float* values  = (const float*)d_in[2];
  // d_in[3] = k (constant 32 per reference)

  __hip_bfloat16* qn = (__hip_bfloat16*)d_ws;                 // 2 MB
  __hip_bfloat16* kn = qn + (size_t)B_Q * DIM;                // 102.4 MB (L2-resident)

  float* out_combined = (float*)d_out;
  float* out_topk     = out_combined + (size_t)B_Q * V_DIM;

  bnm_normalize_bf16<<<B_Q, 256, 0, stream>>>(queries, qn);
  bnm_normalize_bf16<<<N_K, 256, 0, stream>>>(keys, kn);
  bnm_retrieve<<<B_Q / TQ, 256, 0, stream>>>(qn, kn, values, out_combined, out_topk);
}
